// CDVAE_7275674599866
// MI455X (gfx1250) — matrix operand order repaired and checked by reading, not compiled, not run on hardware
//
#include <hip/hip_runtime.h>

// ---------------- problem constants ----------------
#define BB_   16
#define NPS_  64
#define HID_  512
#define LAY_  6
#define NSPEC_ 100
#define EDIM_ 64
#define NN_   1024   // BB_*NPS_

typedef __bf16 bf16_t;
typedef __attribute__((ext_vector_type(16))) __bf16 v16bf;
typedef __attribute__((ext_vector_type(8)))  __bf16 v8bf;
typedef __attribute__((ext_vector_type(8)))  float  v8f;

union FragU { v16bf v; v8bf h[2]; };

// A fragment: 16-bit A 16x32 layout. lane<16 holds K {k0..k0+7, k0+16..k0+23},
// lane>=16 holds {k0+8..k0+15, k0+24..k0+31}. Caller passes p = row_base + k0 + half*8.
__device__ __forceinline__ v16bf load_fragA(const bf16_t* p) {
  FragU f;
  f.h[0] = *(const v8bf*)(p);
  f.h[1] = *(const v8bf*)(p + 16);
  return f.v;
}
// B fragment: packed contiguous 16 bf16 per lane.
__device__ __forceinline__ v16bf load_fragB(const bf16_t* p) {
  FragU f;
  const v8bf* q = (const v8bf*)p;
  f.h[0] = q[0];
  f.h[1] = q[1];
  return f.v;
}

__device__ __forceinline__ v8f wmma_bf(v16bf a, v16bf b, v8f c) {
  return __builtin_amdgcn_wmma_f32_16x16x32_bf16(false, a, false, b, (short)0, c, false, false);
}

__device__ __forceinline__ float silu_f(float x) { return x / (1.0f + __expf(-x)); }

// force a value into an SGPR (wave-uniform by construction)
__device__ __forceinline__ int uni(int x) { return __builtin_amdgcn_readfirstlane(x); }

// ---------------- weight pack: f32 [K x Nsrc] -> bf16 fragment layout [kt][nt][lane][16]
// B-layout: lane l -> column n = nt*16 + (l&15); element s -> k = kt*32 + (l>>4)*8 + (s&7) + (s>>3)*16.
__global__ void pack_w(const float* __restrict__ W, bf16_t* __restrict__ out,
                       int K, int Nsrc, int Npad) {
  int gid = blockIdx.x * blockDim.x + threadIdx.x;
  int ntp = Npad >> 4;
  int total = (K >> 5) * ntp * 512;
  if (gid >= total) return;
  int s    = gid & 15;
  int lane = (gid >> 4) & 31;
  int tile = gid >> 9;            // kt*ntp + nt
  int nt = tile % ntp, kt = tile / ntp;
  int n = nt * 16 + (lane & 15);
  int k = kt * 32 + (lane >> 4) * 8 + (s & 7) + (s >> 3) * 16;
  float v = (n < Nsrc) ? W[(size_t)k * Nsrc + n] : 0.0f;
  out[gid] = (bf16_t)v;
}

// ---------------- Wec = We2 @ Wc1 per layer (f32 scratch, later packed) ----------------
__global__ void wec_kernel(const float* __restrict__ We2, const float* __restrict__ Wc1,
                           float* __restrict__ out) {
  int gid = blockIdx.x * blockDim.x + threadIdx.x;   // 6*512*512
  int n = gid & 511;
  int k = (gid >> 9) & 511;
  int l = gid >> 18;
  const float* a = We2 + ((size_t)l * 512 + k) * 512;
  const float* b = Wc1 + (size_t)l * 512 * 512 + n;
  float acc = 0.0f;
  for (int r = 0; r < 512; ++r) acc += a[r] * b[(size_t)r * 512];
  out[gid] = acc;
}

// bvec = be2 @ Wc1 + bc1   ([6,512])
__global__ void bvec_kernel(const float* __restrict__ be2, const float* __restrict__ Wc1,
                            const float* __restrict__ bc1, float* __restrict__ out) {
  int gid = blockIdx.x * blockDim.x + threadIdx.x;   // 6*512
  if (gid >= LAY_ * 512) return;
  int n = gid & 511, l = gid >> 9;
  float acc = bc1[gid];
  for (int r = 0; r < 512; ++r) acc += be2[l * 512 + r] * Wc1[((size_t)l * 512 + r) * 512 + n];
  out[gid] = acc;
}

// ---------------- time embedding: te[B,512] ----------------
__global__ __launch_bounds__(256) void tembed_kernel(
    const float* __restrict__ t, const float* __restrict__ tW,
    const float* __restrict__ w1, const float* __restrict__ b1,
    const float* __restrict__ w2, const float* __restrict__ b2,
    float* __restrict__ te) {
  __shared__ float g[512];
  __shared__ float hbuf[2048];
  int b = blockIdx.x, tid = threadIdx.x;
  if (tid < 256) {
    float p = t[b] * tW[tid] * 6.283185307179586f;
    float s, c;
    __sincosf(p, &s, &c);
    g[tid] = s; g[256 + tid] = c;
  }
  __syncthreads();
  for (int o = tid; o < 2048; o += 256) {
    float acc = b1[o];
    for (int k = 0; k < 512; ++k) acc += g[k] * w1[(size_t)k * 2048 + o];
    hbuf[o] = silu_f(acc);
  }
  __syncthreads();
  for (int o = tid; o < 512; o += 256) {
    float acc = b2[o];
    for (int k = 0; k < 2048; ++k) acc += hbuf[k] * w2[(size_t)k * 512 + o];
    te[(size_t)b * 512 + o] = acc;
  }
}

// ---------------- node init: h_bf = bf16(atom_tab[type] + te[batch]) ----------------
__global__ void node_init(const int* __restrict__ types, const int* __restrict__ batch,
                          const float* __restrict__ atab, const float* __restrict__ te,
                          bf16_t* __restrict__ h_bf) {
  int idx = blockIdx.x * blockDim.x + threadIdx.x;   // NN_*512
  int n = idx >> 9, f = idx & 511;
  h_bf[idx] = (bf16_t)(atab[(size_t)types[n] * 512 + f] + te[(size_t)batch[n] * 512 + f]);
}

__global__ void copy_c(const float* __restrict__ src, float* __restrict__ dst) {
  int i = blockIdx.x * blockDim.x + threadIdx.x;
  if (i < NN_ * 3) dst[i] = src[i];
}
__global__ void add_cu(float* __restrict__ c, const float* __restrict__ cu) {
  int i = blockIdx.x * blockDim.x + threadIdx.x;
  if (i < NN_ * 3) c[i] += cu[i];
}

// ---------------- edge features: mask + ea (bf16) from initial coords ----------------
__global__ __launch_bounds__(64) void edge_feat(
    const float* __restrict__ coords, const float* __restrict__ eW,
    const float* __restrict__ w1, const float* __restrict__ b1,
    const float* __restrict__ w2, const float* __restrict__ b2,
    float* __restrict__ maskg, bf16_t* __restrict__ eag) {
  int node = blockIdx.x;             // b*64 + i
  int bb = node >> 6, i = node & 63;
  int j = threadIdx.x;
  int nj = (bb << 6) + j;
  float dx = coords[node * 3 + 0] - coords[nj * 3 + 0];
  float dy = coords[node * 3 + 1] - coords[nj * 3 + 1];
  float dz = coords[node * 3 + 2] - coords[nj * 3 + 2];
  float d0 = sqrtf(dx * dx + dy * dy + dz * dz + 1e-12f);
  float mk = (d0 < 8.0f && j != i) ? 1.0f : 0.0f;
  maskg[(size_t)node * 64 + j] = mk;
  float g[64];
  for (int w = 0; w < 32; ++w) {
    float p = d0 * eW[w] * 6.283185307179586f;
    float s, c;
    __sincosf(p, &s, &c);
    g[w] = s; g[32 + w] = c;
  }
  float h1[64];
  for (int o = 0; o < 64; ++o) {
    float acc = b1[o];
    for (int k = 0; k < 64; ++k) acc += g[k] * w1[k * 64 + o];
    h1[o] = silu_f(acc);
  }
  bf16_t* dst = eag + (size_t)node * 4096 + (size_t)j * 64;
  for (int o = 0; o < 64; ++o) {
    float acc = b2[o];
    for (int k = 0; k < 64; ++k) acc += h1[k] * w2[k * 64 + o];
    dst[o] = (bf16_t)acc;
  }
}

// ---------------- generic bf16 WMMA GEMM ----------------
// out = act( A1@B1 [+ A2@B2] + bias + rowScale[row]*rowVec[col] )
// A row-major bf16 (lda=K), B packed fragment layout.
// Block = 256 threads = 8 waves; wave -> 1 M-tile x 4 N-tiles; block tile 64 x 128.
__global__ __launch_bounds__(256) void gemm_bf16(
    const bf16_t* __restrict__ A1, const bf16_t* __restrict__ B1, int K1,
    const bf16_t* __restrict__ A2, const bf16_t* __restrict__ B2, int K2,
    const float* __restrict__ bias,
    const float* __restrict__ rowScale, const float* __restrict__ rowVec,
    int Npad, int Nreal, int act,
    float* __restrict__ outF, int outStride,
    bf16_t* __restrict__ outB) {
  const int tid  = threadIdx.x;
  const int lane = tid & 31, lrow = lane & 15, half = lane >> 4;
  const int wv   = uni(tid >> 5);            // wave-uniform in SGPR
  const int mt   = wv >> 1;
  const int row0 = blockIdx.x * 64;
  const int ntp  = Npad >> 4;
  const int nt0  = blockIdx.y * 8 + (wv & 1) * 4;
  const bool full = (nt0 + 4) <= ntp;        // scalar

  v8f acc[4];
#pragma unroll
  for (int u = 0; u < 4; ++u) { v8f z = {0.f,0.f,0.f,0.f,0.f,0.f,0.f,0.f}; acc[u] = z; }

  const int arow = row0 + mt * 16 + lrow;

  if (full) {
    {
      const bf16_t* ap = A1 + (size_t)arow * K1 + half * 8;
      const bf16_t* bp = B1 + ((size_t)nt0 * 32 + lane) * 16;
      const int kts = K1 >> 5;
      const size_t bstep = (size_t)ntp * 512;     // elements per kt
#pragma unroll 4
      for (int kt = 0; kt < kts; ++kt) {
        v16bf a = load_fragA(ap + kt * 32);
        const bf16_t* bk = bp + (size_t)kt * bstep;
#pragma unroll
        for (int u = 0; u < 4; ++u)
          acc[u] = wmma_bf(a, load_fragB(bk + u * 512), acc[u]);
      }
    }
    if (A2 != nullptr) {
      const bf16_t* ap = A2 + (size_t)arow * K2 + half * 8;
      const bf16_t* bp = B2 + ((size_t)nt0 * 32 + lane) * 16;
      const int kts = K2 >> 5;
      const size_t bstep = (size_t)ntp * 512;
#pragma unroll 4
      for (int kt = 0; kt < kts; ++kt) {
        v16bf a = load_fragA(ap + kt * 32);
        const bf16_t* bk = bp + (size_t)kt * bstep;
#pragma unroll
        for (int u = 0; u < 4; ++u)
          acc[u] = wmma_bf(a, load_fragB(bk + u * 512), acc[u]);
      }
    }
  } else {
    const int nv = ntp - nt0;                  // scalar; may be <= 0
    {
      const bf16_t* ap = A1 + (size_t)arow * K1 + half * 8;
      const int kts = K1 >> 5;
      for (int kt = 0; kt < kts; ++kt) {
        v16bf a = load_fragA(ap + kt * 32);
        for (int u = 0; u < 4; ++u)
          if (u < nv)
            acc[u] = wmma_bf(a, load_fragB(B1 + ((size_t)(kt * ntp + nt0 + u) * 32 + lane) * 16), acc[u]);
      }
    }
    if (A2 != nullptr) {
      const bf16_t* ap = A2 + (size_t)arow * K2 + half * 8;
      const int kts = K2 >> 5;
      for (int kt = 0; kt < kts; ++kt) {
        v16bf a = load_fragA(ap + kt * 32);
        for (int u = 0; u < 4; ++u)
          if (u < nv)
            acc[u] = wmma_bf(a, load_fragB(B2 + ((size_t)(kt * ntp + nt0 + u) * 32 + lane) * 16), acc[u]);
      }
    }
  }

  // epilogue
#pragma unroll
  for (int u = 0; u < 4; ++u) {
    int nt = nt0 + u;
    if (nt >= ntp) continue;                   // scalar guard
    int col = nt * 16 + lrow;
    float add = (bias && col < Nreal) ? bias[col] : 0.0f;
    float rv = (rowVec && col < Nreal) ? rowVec[col] : 0.0f;
#pragma unroll
    for (int v = 0; v < 8; ++v) {
      int row = row0 + mt * 16 + v + 8 * half;
      float x = acc[u][v] + add;
      if (rowScale) x += rowScale[row] * rv;
      if (act) x = silu_f(x);
      if (outF && col < Nreal) outF[(size_t)row * outStride + col] = x;
      if (outB) outB[(size_t)row * Npad + col] = (bf16_t)x;
    }
  }
}

// ---------------- fused per-(b,i) edge kernel ----------------
// phase1: C1 = ea@W1c + P[i] + Q[b,j]  (P holds h@W1a+be1, Q holds h@W1b)
//         s = SiLU(C1) -> LDS; masked column sums -> Spre
// phase2: t = s@Wec + bvec ; cs = Wc2^T SiLU(t) ; cu = sum_j mask*cs*(ci-cj)/dist
__global__ __launch_bounds__(256) void edge_layer(
    const bf16_t* __restrict__ eag, const bf16_t* __restrict__ W1c_p,
    const float* __restrict__ Pg, const float* __restrict__ Qg,
    const bf16_t* __restrict__ Wec_p, const float* __restrict__ bvec,
    const float* __restrict__ maskg, const float* __restrict__ cc,
    const float* __restrict__ Wc2, const float* __restrict__ bc2p,
    bf16_t* __restrict__ SpreB, float* __restrict__ cntg, float* __restrict__ cug) {
  __shared__ bf16_t sS[64][520];   // SiLU(C1), padded stride for LDS banking
  __shared__ float sCol[512];
  __shared__ float sCs[64];
  __shared__ float sMask[64];
  __shared__ float sCu[3];

  const int tid  = threadIdx.x;
  const int lane = tid & 31, lrow = lane & 15, half = lane >> 4;
  const int wv   = uni(tid >> 5);            // wave-uniform in SGPR
  const int mt   = wv >> 1, ntB = (wv & 1) * 16;
  const int node = blockIdx.x, bb = node >> 6;

  if (tid < 512) sCol[tid] = 0.0f;
  if (tid < 64) { sCs[tid] = 0.0f; sMask[tid] = maskg[(size_t)node * 64 + tid]; }
  if (tid < 3)  sCu[tid] = 0.0f;
  __syncthreads();

  v8f acc[16];
#pragma unroll
  for (int t = 0; t < 16; ++t) { v8f z = {0.f,0.f,0.f,0.f,0.f,0.f,0.f,0.f}; acc[t] = z; }

  // ---- phase 1: ea[64x64] @ W1c[64x512] ----
  const bf16_t* Aea = eag + (size_t)node * 4096 + (size_t)(mt * 16 + lrow) * 64 + half * 8;
  const bf16_t* Bp1 = W1c_p + ((size_t)ntB * 32 + lane) * 16;
#pragma unroll
  for (int kt = 0; kt < 2; ++kt) {
    v16bf a = load_fragA(Aea + kt * 32);
    const bf16_t* bk = Bp1 + (size_t)kt * 32 * 512;
#pragma unroll
    for (int t = 0; t < 16; ++t)
      acc[t] = wmma_bf(a, load_fragB(bk + (size_t)t * 512), acc[t]);
  }
#pragma unroll
  for (int t = 0; t < 16; ++t) {
    int col = (ntB + t) * 16 + lrow;
    float pv = Pg[(size_t)node * 512 + col];
    float msum = 0.0f;
#pragma unroll
    for (int v = 0; v < 8; ++v) {
      int row = mt * 16 + v + 8 * half;
      float x = acc[t][v] + pv + Qg[((size_t)((bb << 6) + row)) * 512 + col];
      float s = silu_f(x);
      sS[row][col] = (bf16_t)s;
      msum += s * sMask[row];
    }
    atomicAdd(&sCol[col], msum);
  }
  __syncthreads();

  // ---- phase 2: t = s[64x512] @ Wec[512x512] + bvec ----
#pragma unroll
  for (int t = 0; t < 16; ++t) {
    float bv = bvec[(ntB + t) * 16 + lrow];
    v8f z;
#pragma unroll
    for (int v = 0; v < 8; ++v) z[v] = bv;
    acc[t] = z;
  }
  {
    const bf16_t* Bp2 = Wec_p + ((size_t)ntB * 32 + lane) * 16;
#pragma unroll 2
    for (int kt = 0; kt < 16; ++kt) {
      v16bf a = load_fragA(&sS[mt * 16 + lrow][kt * 32 + half * 8]);
      const bf16_t* bk = Bp2 + (size_t)kt * 32 * 512;
#pragma unroll
      for (int t = 0; t < 16; ++t)
        acc[t] = wmma_bf(a, load_fragB(bk + (size_t)t * 512), acc[t]);
    }
  }
  {
    float rp[8] = {0.f, 0.f, 0.f, 0.f, 0.f, 0.f, 0.f, 0.f};
#pragma unroll
    for (int t = 0; t < 16; ++t) {
      int col = (ntB + t) * 16 + lrow;
      float w2 = Wc2[col];
#pragma unroll
      for (int v = 0; v < 8; ++v) rp[v] += silu_f(acc[t][v]) * w2;
    }
#pragma unroll
    for (int v = 0; v < 8; ++v) atomicAdd(&sCs[mt * 16 + v + 8 * half], rp[v]);
  }
  __syncthreads();

  // ---- tail: Spre, cnt, cu ----
  for (int f = tid; f < 512; f += 256) SpreB[(size_t)node * 512 + f] = (bf16_t)sCol[f];
  if (tid == 0) {
    float cn = 0.0f;
    for (int j = 0; j < 64; ++j) cn += sMask[j];
    cntg[node] = cn;
  }
  if (tid < 64) {
    float mk = sMask[tid];
    if (mk > 0.0f) {
      int nj = (bb << 6) + tid;
      float dx = cc[node * 3 + 0] - cc[nj * 3 + 0];
      float dy = cc[node * 3 + 1] - cc[nj * 3 + 1];
      float dz = cc[node * 3 + 2] - cc[nj * 3 + 2];
      float dist = sqrtf(dx * dx + dy * dy + dz * dz + 1e-12f) + 1e-8f;
      float s = (sCs[tid] + bc2p[0]) * mk / dist;
      atomicAdd(&sCu[0], s * dx);
      atomicAdd(&sCu[1], s * dy);
      atomicAdd(&sCu[2], s * dz);
    }
  }
  __syncthreads();
  if (tid < 3) cug[(size_t)node * 3 + tid] = sCu[tid];
}

// ---------------- host side ----------------
extern "C" void kernel_launch(void* const* d_in, const int* in_sizes, int n_in,
                              void* d_out, int out_size, void* d_ws, size_t ws_size,
                              hipStream_t stream) {
  (void)in_sizes; (void)n_in; (void)out_size; (void)ws_size;
  const float* coords = (const float*)d_in[0];
  const int*   atom_types = (const int*)d_in[1];
  const float* t      = (const float*)d_in[2];
  const int*   batch  = (const int*)d_in[3];
  const float* atab   = (const float*)d_in[4];
  const float* tW     = (const float*)d_in[5];
  const float* t_w1   = (const float*)d_in[6];
  const float* t_b1   = (const float*)d_in[7];
  const float* t_w2   = (const float*)d_in[8];
  const float* t_b2   = (const float*)d_in[9];
  const float* eW     = (const float*)d_in[10];
  const float* e_w1   = (const float*)d_in[11];
  const float* e_b1   = (const float*)d_in[12];
  const float* e_w2   = (const float*)d_in[13];
  const float* e_b2   = (const float*)d_in[14];
  const float* We1    = (const float*)d_in[15];
  const float* be1    = (const float*)d_in[16];
  const float* We2    = (const float*)d_in[17];
  const float* be2    = (const float*)d_in[18];
  const float* Wn1    = (const float*)d_in[19];
  const float* bn1    = (const float*)d_in[20];
  const float* Wn2    = (const float*)d_in[21];
  const float* bn2    = (const float*)d_in[22];
  const float* Wc1    = (const float*)d_in[23];
  const float* bc1    = (const float*)d_in[24];
  const float* Wc2    = (const float*)d_in[25];
  const float* bc2    = (const float*)d_in[26];
  const float* Wp1    = (const float*)d_in[27];
  const float* bp1    = (const float*)d_in[28];
  const float* Wp2    = (const float*)d_in[29];
  const float* bp2    = (const float*)d_in[30];
  const float* Wq1    = (const float*)d_in[31];
  const float* bq1    = (const float*)d_in[32];
  const float* Wq2    = (const float*)d_in[33];
  const float* bq2    = (const float*)d_in[34];

  char* ws = (char*)d_ws;
  size_t off = 0;
  auto alloc = [&](size_t bytes) -> char* {
    char* p = ws + off;
    off += (bytes + 255) & ~(size_t)255;
    return p;
  };
  const size_t W512 = (size_t)512 * 512;      // elements per 512x512 matrix
  bf16_t* W1a_p  = (bf16_t*)alloc(LAY_ * W512 * 2);
  bf16_t* W1b_p  = (bf16_t*)alloc(LAY_ * W512 * 2);
  bf16_t* W1c_p  = (bf16_t*)alloc(LAY_ * (size_t)64 * 512 * 2);
  bf16_t* We2_p  = (bf16_t*)alloc(LAY_ * W512 * 2);
  bf16_t* Wn1a_p = (bf16_t*)alloc(LAY_ * W512 * 2);
  bf16_t* Wn1b_p = (bf16_t*)alloc(LAY_ * W512 * 2);
  bf16_t* Wn2_p  = (bf16_t*)alloc(LAY_ * W512 * 2);
  bf16_t* Wec_p  = (bf16_t*)alloc(LAY_ * W512 * 2);
  bf16_t* Wp1_p  = (bf16_t*)alloc(W512 * 2);
  bf16_t* Wq1_p  = (bf16_t*)alloc(W512 * 2);
  bf16_t* Wp2_p  = (bf16_t*)alloc((size_t)512 * 16 * 2);
  bf16_t* Wq2_p  = (bf16_t*)alloc((size_t)512 * 112 * 2);
  float*  Wec_f  = (float*)alloc(LAY_ * W512 * 4);
  float*  bvecg  = (float*)alloc(LAY_ * 512 * 4);
  float*  te     = (float*)alloc((size_t)BB_ * 512 * 4);
  bf16_t* h_bf   = (bf16_t*)alloc((size_t)NN_ * 512 * 2);
  bf16_t* ea_bf  = (bf16_t*)alloc((size_t)NN_ * 64 * 64 * 2);
  float*  maskg  = (float*)alloc((size_t)NN_ * 64 * 4);
  float*  c_ws   = (float*)alloc((size_t)NN_ * 3 * 4);
  float*  Pg     = (float*)alloc((size_t)NN_ * 512 * 4);
  float*  Qg     = (float*)alloc((size_t)NN_ * 512 * 4);
  bf16_t* SpreB  = (bf16_t*)alloc((size_t)NN_ * 512 * 2);
  float*  cntg   = (float*)alloc((size_t)NN_ * 4);
  float*  cug    = (float*)alloc((size_t)NN_ * 3 * 4);
  bf16_t* NM_bf  = (bf16_t*)alloc((size_t)NN_ * 512 * 2);
  bf16_t* U_bf   = (bf16_t*)alloc((size_t)NN_ * 512 * 2);
  bf16_t* hp_bf  = (bf16_t*)alloc((size_t)NN_ * 512 * 2);
  bf16_t* hq_bf  = (bf16_t*)alloc((size_t)NN_ * 512 * 2);

  auto pack = [&](const float* src, bf16_t* dst, int K, int Nsrc, int Npad) {
    int total = (K / 32) * (Npad / 16) * 512;
    pack_w<<<(total + 255) / 256, 256, 0, stream>>>(src, dst, K, Nsrc, Npad);
  };

  // ---- weight prep ----
  for (int l = 0; l < LAY_; ++l) {
    const float* We1l = We1 + (size_t)l * 1088 * 512;
    pack(We1l,              W1a_p + l * W512,          512, 512, 512);
    pack(We1l + 512 * 512,  W1b_p + l * W512,          512, 512, 512);
    pack(We1l + 1024 * 512, W1c_p + (size_t)l * 64 * 512, 64, 512, 512);
    pack(We2 + l * W512,            We2_p  + l * W512, 512, 512, 512);
    pack(Wn1 + (size_t)l * 1024 * 512,             Wn1a_p + l * W512, 512, 512, 512);
    pack(Wn1 + (size_t)l * 1024 * 512 + 512 * 512, Wn1b_p + l * W512, 512, 512, 512);
    pack(Wn2 + l * W512,            Wn2_p  + l * W512, 512, 512, 512);
  }
  wec_kernel<<<(LAY_ * 512 * 512) / 256, 256, 0, stream>>>(We2, Wc1, Wec_f);
  bvec_kernel<<<(LAY_ * 512 + 255) / 256, 256, 0, stream>>>(be2, Wc1, bc1, bvecg);
  for (int l = 0; l < LAY_; ++l)
    pack(Wec_f + l * W512, Wec_p + l * W512, 512, 512, 512);
  pack(Wp1, Wp1_p, 512, 512, 512);
  pack(Wq1, Wq1_p, 512, 512, 512);
  pack(Wp2, Wp2_p, 512, 3, 16);
  pack(Wq2, Wq2_p, 512, 100, 112);

  // ---- embeddings / graph setup ----
  tembed_kernel<<<BB_, 256, 0, stream>>>(t, tW, t_w1, t_b1, t_w2, t_b2, te);
  node_init<<<(NN_ * 512) / 256, 256, 0, stream>>>(atom_types, batch, atab, te, h_bf);
  copy_c<<<(NN_ * 3 + 255) / 256, 256, 0, stream>>>(coords, c_ws);
  edge_feat<<<NN_, 64, 0, stream>>>(coords, eW, e_w1, e_b1, e_w2, e_b2, maskg, ea_bf);

  dim3 g512(NN_ / 64, 512 / 128);
  // ---- layers ----
  for (int l = 0; l < LAY_; ++l) {
    // P = h@W1a + be1 ; Q = h@W1b
    gemm_bf16<<<g512, 256, 0, stream>>>(h_bf, W1a_p + l * W512, 512,
                                        nullptr, nullptr, 0,
                                        be1 + l * 512, nullptr, nullptr,
                                        512, 512, 0, Pg, 512, nullptr);
    gemm_bf16<<<g512, 256, 0, stream>>>(h_bf, W1b_p + l * W512, 512,
                                        nullptr, nullptr, 0,
                                        nullptr, nullptr, nullptr,
                                        512, 512, 0, Qg, 512, nullptr);
    edge_layer<<<NN_, 256, 0, stream>>>(ea_bf, W1c_p + (size_t)l * 64 * 512,
                                        Pg, Qg, Wec_p + l * W512, bvecg + l * 512,
                                        maskg, c_ws, Wc2 + l * 512, bc2 + l,
                                        SpreB, cntg, cug);
    // NM = Spre@We2 + cnt*be2
    gemm_bf16<<<g512, 256, 0, stream>>>(SpreB, We2_p + l * W512, 512,
                                        nullptr, nullptr, 0,
                                        nullptr, cntg, be2 + l * 512,
                                        512, 512, 0, nullptr, 0, NM_bf);
    // U = SiLU(h@Wn1a + NM@Wn1b + bn1)
    gemm_bf16<<<g512, 256, 0, stream>>>(h_bf, Wn1a_p + l * W512, 512,
                                        NM_bf, Wn1b_p + l * W512, 512,
                                        bn1 + l * 512, nullptr, nullptr,
                                        512, 512, 1, nullptr, 0, U_bf);
    // h = U@Wn2 + bn2
    gemm_bf16<<<g512, 256, 0, stream>>>(U_bf, Wn2_p + l * W512, 512,
                                        nullptr, nullptr, 0,
                                        bn2 + l * 512, nullptr, nullptr,
                                        512, 512, 0, nullptr, 0, h_bf);
    add_cu<<<(NN_ * 3 + 255) / 256, 256, 0, stream>>>(c_ws, cug);
  }

  // ---- output heads ----
  float* outF = (float*)d_out;
  gemm_bf16<<<g512, 256, 0, stream>>>(h_bf, Wp1_p, 512, nullptr, nullptr, 0,
                                      bp1, nullptr, nullptr, 512, 512, 1,
                                      nullptr, 0, hp_bf);
  dim3 g16(NN_ / 64, 1);
  gemm_bf16<<<g16, 256, 0, stream>>>(hp_bf, Wp2_p, 512, nullptr, nullptr, 0,
                                     bp2, nullptr, nullptr, 16, 3, 0,
                                     outF, 3, nullptr);
  gemm_bf16<<<g512, 256, 0, stream>>>(h_bf, Wq1_p, 512, nullptr, nullptr, 0,
                                      bq1, nullptr, nullptr, 512, 512, 1,
                                      nullptr, 0, hq_bf);
  dim3 g112(NN_ / 64, 1);
  gemm_bf16<<<g112, 256, 0, stream>>>(hq_bf, Wq2_p, 512, nullptr, nullptr, 0,
                                      bq2, nullptr, nullptr, 112, 100, 0,
                                      outF + (size_t)NN_ * 3, 100, nullptr);
}
